// HGT_14508399526531
// MI455X (gfx1250) — compile-verified, matrix operand order checked by reading, not run on hardware
//
#include <hip/hip_runtime.h>
#include <math.h>

typedef __attribute__((ext_vector_type(16))) __bf16 v16bf;
typedef __attribute__((ext_vector_type(8)))  float  v8f;
typedef __attribute__((ext_vector_type(2)))  float  v2f;
typedef unsigned int u32x4 __attribute__((ext_vector_type(4)));
typedef int          i32x8 __attribute__((ext_vector_type(8)));
typedef int          i32x4 __attribute__((ext_vector_type(4)));

#ifndef __has_builtin
#define __has_builtin(x) 0
#endif
#if __has_builtin(__builtin_amdgcn_tensor_load_to_lds)
#define HAVE_TDM 1
#else
#define HAVE_TDM 0
#endif

// ---------------------------------------------------------------------------
// GEMM: Y[N,128] = (gelu?)(X[N,128]) @ W[128,128] + b ; optional sigmoid-gated
// skip blend with xres. 256 threads = 8 waves. Block computes 64 rows x 128
// cols. A tile staged in LDS by the Tensor Data Mover (zero-fill past N via
// tensor_dim1 = N-row0); wave w owns cols [16w,16w+16) and iterates 4 M-tiles
// reusing its B fragment (16 bf16 WMMAs / wave / block).
// ---------------------------------------------------------------------------
__global__ __launch_bounds__(256) void gemm128_kernel(
    const float* __restrict__ X, const float* __restrict__ W,
    const float* __restrict__ bias, float* __restrict__ Y,
    const float* __restrict__ xres, const float* __restrict__ skipgate,
    int N, int apply_gelu)
{
  __shared__ float As[64][128];                  // 32 KB A tile (f32)
  const int tid  = threadIdx.x;
  const int wave = tid >> 5;
  const int lane = tid & 31;
  const int row0 = blockIdx.x * 64;

#if HAVE_TDM
  if (wave == 0) {                               // one wave drives the TDM
    unsigned long long ga = (unsigned long long)(const void*)(X + (size_t)row0 * 128);
    unsigned lds = (unsigned)(unsigned long long)(const void*)&As[0][0];
    unsigned rows = (unsigned)(N - row0);        // OOB rows zero-filled by TDM
    u32x4 g0;
    g0[0] = 1u;                                  // count=1 (user descriptor)
    g0[1] = lds;                                 // D#.lds_addr
    g0[2] = (unsigned)ga;                        // global_addr[31:0]
    g0[3] = (unsigned)((ga >> 32) & 0x01FFFFFFull) | 0x80000000u;  // [56:32] | type=2
    i32x8 g1;
    g1[0] = (int)(2u << 16);                     // data_size = 4 bytes
    g1[1] = (int)(128u << 16);                   // tensor_dim0 = 128
    g1[2] = (int)((rows & 0xFFFFu) << 16);       // tensor_dim1 lo
    g1[3] = (int)(((rows >> 16) & 0xFFFFu) | (128u << 16)); // dim1 hi | tile_dim0=128
    g1[4] = 64;                                  // tile_dim1 = 64
    g1[5] = 128;                                 // tensor_dim0_stride = 128
    g1[6] = 0;
    g1[7] = 0;
    i32x4 z4 = {0, 0, 0, 0};
#if defined(__clang_major__) && (__clang_major__ >= 23)
    i32x8 z8 = {0, 0, 0, 0, 0, 0, 0, 0};
    __builtin_amdgcn_tensor_load_to_lds(g0, g1, z4, z4, z8, 0);
#else
    __builtin_amdgcn_tensor_load_to_lds(g0, g1, z4, z4, 0);
#endif
#if __has_builtin(__builtin_amdgcn_s_wait_tensorcnt)
    __builtin_amdgcn_s_wait_tensorcnt(0);
#else
    asm volatile("s_wait_tensorcnt 0x0" ::: "memory");
#endif
  }
  __syncthreads();
#else
  for (int i = tid; i < 64 * 128; i += 256) {    // fallback cooperative stage
    int r = i >> 7, c = i & 127;
    int gr = row0 + r;
    (&As[0][0])[i] = (gr < N) ? X[(size_t)gr * 128 + c] : 0.f;
  }
  __syncthreads();
#endif

  if (apply_gelu) {                              // exact GELU in-place on tile
    for (int i = tid; i < 64 * 128; i += 256) {
      float v = (&As[0][0])[i];
      (&As[0][0])[i] = 0.5f * v * (1.f + erff(v * 0.70710678118654752f));
    }
    __syncthreads();
  }

  const int kh   = (lane >> 4) << 3;             // 0 / 8 (K and M half select)
  const int colL = wave * 16 + (lane & 15);
  v8f acc[4] = {{}, {}, {}, {}};
  #pragma unroll
  for (int kk = 0; kk < 4; ++kk) {               // K = 128 = 4 * 32
    v16bf b;
    #pragma unroll
    for (int j = 0; j < 8; ++j) {                // ISA B layout (bf16 32x16)
      int k = kk * 32 + ((j & 4) << 2) + ((j & 3) << 1) + kh;
      b[2 * j]     = (__bf16)W[(size_t)k * 128 + colL];
      b[2 * j + 1] = (__bf16)W[(size_t)(k + 1) * 128 + colL];
    }
    #pragma unroll
    for (int mt = 0; mt < 4; ++mt) {             // 4 M-tiles reuse B fragment
      v16bf a;
      #pragma unroll
      for (int j = 0; j < 8; ++j) {              // ISA A layout (bf16 16x32)
        int k = kk * 32 + ((j & 4) << 2) + ((j & 3) << 1) + kh;
        int r = mt * 16 + (lane & 15);
        a[2 * j]     = (__bf16)As[r][k];
        a[2 * j + 1] = (__bf16)As[r][k + 1];
      }
      acc[mt] = __builtin_amdgcn_wmma_f32_16x16x32_bf16(false, a, false, b,
                                                        (short)0, acc[mt],
                                                        false, false);
    }
  }

  float g0s = 1.f, g1s = 0.f;
  if (skipgate) { float s = skipgate[0]; g0s = 1.f / (1.f + __expf(-s)); g1s = 1.f - g0s; }
  #pragma unroll
  for (int mt = 0; mt < 4; ++mt) {
    #pragma unroll
    for (int j = 0; j < 8; ++j) {                // D: VGPR j -> row j (+8 hi)
      int r = row0 + mt * 16 + j + kh;
      if (r < N) {
        float v = acc[mt][j] + bias[colL];
        if (xres) v = g0s * v + g1s * xres[(size_t)r * 128 + colL];
        Y[(size_t)r * 128 + colL] = v;
      }
    }
  }
}

// ---------------------------------------------------------------------------
// Per-relation head transform: O[n,h,:] = S[n,h,:] @ R[h]  (H=8, D=16).
// One wave per (16-node tile, head). Exact fp32 via V_WMMA_F32_16X16X4_F32.
// ---------------------------------------------------------------------------
__global__ __launch_bounds__(32) void reltrans_kernel(
    const float* __restrict__ S, const float* __restrict__ R,
    float* __restrict__ O, int Ns)
{
  const int lane  = threadIdx.x;
  const int h     = blockIdx.y;
  const int row0  = blockIdx.x * 16;
  const int r     = row0 + (lane & 15);
  const int c     = lane & 15;
  const int khalf = (lane >> 4) << 1;            // 0 / 2 (K half select)
  v8f acc = {};
  #pragma unroll
  for (int kk = 0; kk < 4; ++kk) {               // K = 16 = 4 * 4
    int k0 = kk * 4 + khalf;
    v2f a, b;
    a.x = (r < Ns) ? S[(size_t)r * 128 + h * 16 + k0]     : 0.f;
    a.y = (r < Ns) ? S[(size_t)r * 128 + h * 16 + k0 + 1] : 0.f;
    b.x = R[h * 256 + k0 * 16 + c];
    b.y = R[h * 256 + (k0 + 1) * 16 + c];
    acc = __builtin_amdgcn_wmma_f32_16x16x4_f32(false, a, false, b,
                                                (short)0, acc, false, false);
  }
  const int kh = (lane >> 4) << 3;
  #pragma unroll
  for (int j = 0; j < 8; ++j) {
    int rr = row0 + j + kh;
    if (rr < Ns) O[(size_t)rr * 128 + h * 16 + c] = acc[j];
  }
}

// --------------------------- segment softmax ------------------------------
__device__ __forceinline__ unsigned encf(float f) {
  unsigned u = __float_as_uint(f);
  return (u & 0x80000000u) ? ~u : (u | 0x80000000u);     // order-preserving
}
__device__ __forceinline__ float decf(unsigned u) {
  return __uint_as_float((u & 0x80000000u) ? (u ^ 0x80000000u) : ~u);
}

__global__ __launch_bounds__(256) void edge_logits_kernel(
    const int* __restrict__ src, const int* __restrict__ dst,
    const float* __restrict__ Q, const float* __restrict__ KREL,
    const float* __restrict__ prel, float* __restrict__ LOGIT,
    unsigned* __restrict__ MAXB, int E)
{
  int t = blockIdx.x * blockDim.x + threadIdx.x;
  if (t >= E * 8) return;
  int e = t >> 3, h = t & 7;
  if (e + 1024 < E) {                            // hide index-gather latency
    __builtin_prefetch(src + e + 1024, 0, 1);
    __builtin_prefetch(dst + e + 1024, 0, 1);
  }
  int s = src[e], d = dst[e];
  const float* q = Q    + (size_t)d * 128 + h * 16;
  const float* k = KREL + (size_t)s * 128 + h * 16;
  float acc = 0.f;
  #pragma unroll
  for (int i = 0; i < 16; ++i) acc += q[i] * k[i];
  acc *= prel[h] * 0.25f;                        // 1/sqrt(D), D=16
  LOGIT[t] = acc;
  atomicMax(MAXB + (size_t)d * 8 + h, encf(acc));
}

__global__ __launch_bounds__(256) void edge_exp_kernel(
    const int* __restrict__ dst, const float* __restrict__ LOGIT,
    const unsigned* __restrict__ MAXB, float* __restrict__ EXPB,
    float* __restrict__ SUMB, int E)
{
  int t = blockIdx.x * blockDim.x + threadIdx.x;
  if (t >= E * 8) return;
  int e = t >> 3, h = t & 7;
  int d = dst[e];
  float ex = __expf(LOGIT[t] - decf(MAXB[(size_t)d * 8 + h]));
  EXPB[t] = ex;
  atomicAdd(SUMB + (size_t)d * 8 + h, ex);
}

__global__ __launch_bounds__(256) void edge_scatter_kernel(
    const int* __restrict__ src, const int* __restrict__ dst,
    const float* __restrict__ EXPB, const float* __restrict__ SUMB,
    const float* __restrict__ VREL, float* __restrict__ AGG, int E)
{
  int t = blockIdx.x * blockDim.x + threadIdx.x;
  if (t >= E * 8) return;
  int e = t >> 3, h = t & 7;
  int s = src[e], d = dst[e];
  float al = EXPB[t] / (SUMB[(size_t)d * 8 + h] + 1e-16f);
  const float* v = VREL + (size_t)s * 128 + h * 16;
  float*       a = AGG  + (size_t)d * 128 + h * 16;
  #pragma unroll
  for (int i = 0; i < 16; ++i) atomicAdd(a + i, v[i] * al);
}

__global__ __launch_bounds__(128) void gather_kernel(
    const float* __restrict__ X, const int* __restrict__ idx,
    float* __restrict__ out)
{
  out[(size_t)blockIdx.x * 128 + threadIdx.x] =
      X[(size_t)idx[blockIdx.x] * 128 + threadIdx.x];
}

// ---------------------------------------------------------------------------
extern "C" void kernel_launch(void* const* d_in, const int* in_sizes, int n_in,
                              void* d_out, int out_size, void* d_ws, size_t ws_size,
                              hipStream_t stream)
{
  (void)in_sizes; (void)n_in; (void)out_size; (void)ws_size;
  // internal node order: 0=paper 1=author 2=term 3=conference
  const int    NN[4]  = {50000, 30000, 20000, 50};
  const size_t OFF[4] = {0, 50000, 80000, 100000};
  const int    PNT[4] = {2, 0, 3, 1};            // internal -> sorted param idx
  const size_t NTF    = (size_t)100050 * 128;

  float*    ws    = (float*)d_ws;
  float*    Kb    = ws;
  float*    Qb    = ws + NTF;
  float*    Vb    = ws + 2 * NTF;
  float*    AGG   = ws + 3 * NTF;
  float*    X1    = ws + 4 * NTF;
  float*    KREL  = ws + 5 * NTF;                // 50000*128
  float*    VREL  = KREL + (size_t)50000 * 128;
  float*    LOGIT = VREL + (size_t)50000 * 128;  // 100000*8
  float*    EXPB  = LOGIT + 800000;
  unsigned* MAXB  = (unsigned*)(EXPB + 800000);  // 50000*8
  float*    SUMB  = (float*)(MAXB + 400000);
  float*    X2A   = Kb;                          // layer-2 author out (K dead)

  const float* xin[4] = {(const float*)d_in[0], (const float*)d_in[1],
                         (const float*)d_in[2], (const float*)d_in[3]};

  struct ET { int st, dt, E, isrc, idst, pe; };
  const ET ets[6] = {
    {1, 0, 100000,  4,  5, 0},   // author__paper
    {0, 1, 100000,  6,  7, 2},   // paper__author
    {0, 2, 100000,  8,  9, 4},   // paper__term
    {2, 0, 100000, 10, 11, 5},   // term__paper
    {3, 0,  20000, 12, 13, 1},   // conference__paper
    {0, 3,  20000, 14, 15, 3},   // paper__conference
  };

  for (int layer = 0; layer < 2; ++layer) {
    const int base = 17 + 54 * layer;            // 54 pytree leaves per layer
    const float* xl[4];
    for (int t = 0; t < 4; ++t) xl[t] = (layer == 0) ? xin[t] : X1 + OFF[t] * 128;

    // K/Q/V projections (bf16 WMMA, TDM-staged A tiles)
    for (int t = 0; t < 4; ++t) {
      dim3 g((NN[t] + 63) / 64);
      gemm128_kernel<<<g, 256, 0, stream>>>(xl[t], (const float*)d_in[base + 4 + PNT[t]],
          (const float*)d_in[base + 26 + PNT[t]], Kb + OFF[t] * 128, nullptr, nullptr, NN[t], 0);
      gemm128_kernel<<<g, 256, 0, stream>>>(xl[t], (const float*)d_in[base + 8 + PNT[t]],
          (const float*)d_in[base + 30 + PNT[t]], Qb + OFF[t] * 128, nullptr, nullptr, NN[t], 0);
      gemm128_kernel<<<g, 256, 0, stream>>>(xl[t], (const float*)d_in[base + 12 + PNT[t]],
          (const float*)d_in[base + 34 + PNT[t]], Vb + OFF[t] * 128, nullptr, nullptr, NN[t], 0);
    }
    hipMemsetAsync(AGG, 0, NTF * sizeof(float), stream);

    for (int e = 0; e < 6; ++e) {
      const ET& T = ets[e];
      const int Ns = NN[T.st], Nd = NN[T.dt];
      dim3 rg((Ns + 15) / 16, 8);
      reltrans_kernel<<<rg, 32, 0, stream>>>(Kb + OFF[T.st] * 128,
          (const float*)d_in[base + 16 + T.pe], KREL, Ns);   // a_rel
      reltrans_kernel<<<rg, 32, 0, stream>>>(Vb + OFF[T.st] * 128,
          (const float*)d_in[base + 38 + T.pe], VREL, Ns);   // m_rel
      hipMemsetAsync(MAXB, 0, (size_t)Nd * 8 * sizeof(unsigned), stream);
      hipMemsetAsync(SUMB, 0, (size_t)Nd * 8 * sizeof(float), stream);
      const int nt = T.E * 8, nb = (nt + 255) / 256;
      edge_logits_kernel<<<nb, 256, 0, stream>>>((const int*)d_in[T.isrc],
          (const int*)d_in[T.idst], Qb + OFF[T.dt] * 128, KREL,
          (const float*)d_in[base + 44 + T.pe], LOGIT, MAXB, T.E);
      edge_exp_kernel<<<nb, 256, 0, stream>>>((const int*)d_in[T.idst],
          LOGIT, MAXB, EXPB, SUMB, T.E);
      edge_scatter_kernel<<<nb, 256, 0, stream>>>((const int*)d_in[T.isrc],
          (const int*)d_in[T.idst], EXPB, SUMB, VREL, AGG + OFF[T.dt] * 128, T.E);
    }

    // Output projection: gelu(agg) @ W_a + b_a, sigmoid-gated skip.
    if (layer == 0) {
      for (int t = 0; t < 4; ++t) {
        dim3 g((NN[t] + 63) / 64);
        gemm128_kernel<<<g, 256, 0, stream>>>(AGG + OFF[t] * 128,
            (const float*)d_in[base + 0 + PNT[t]], (const float*)d_in[base + 22 + PNT[t]],
            X1 + OFF[t] * 128, xl[t], (const float*)d_in[base + 50 + PNT[t]], NN[t], 1);
      }
    } else {
      // only the author output feeds the final gather
      dim3 g((NN[1] + 63) / 64);
      gemm128_kernel<<<g, 256, 0, stream>>>(AGG + OFF[1] * 128,
          (const float*)d_in[base + 0 + PNT[1]], (const float*)d_in[base + 22 + PNT[1]],
          X2A, xl[1], (const float*)d_in[base + 50 + PNT[1]], NN[1], 1);
    }
  }

  gather_kernel<<<1000, 128, 0, stream>>>(X2A, (const int*)d_in[16], (float*)d_out);
}
